// LSTM4OTDecoder_16123307229677
// MI455X (gfx1250) — 8-bit matrix operand order repaired and checked by reading, compiled, not run on hardware
//
#include <hip/hip_runtime.h>
#include <hip/hip_bf16.h>
#include <cmath>

typedef __attribute__((ext_vector_type(16))) __bf16 v16bf;
typedef __attribute__((ext_vector_type(8)))  float  v8f;
typedef __attribute__((ext_vector_type(16))) int    v16i;

static constexpr int T_LEN = 100;
static constexpr int S_LEN = 400;
static constexpr int B     = 16;
static constexpr int D_IN  = 512;
static constexpr int H     = 512;
static constexpr int CTX   = 1024;
static constexpr int V     = 50000;
static constexpr int G4    = 2048;


__device__ __forceinline__ __bf16 to_bf16(float f) {
  unsigned int u = __float_as_uint(f);
  unsigned int r = (u + 0x7FFFu + ((u >> 16) & 1u)) >> 16;
  unsigned short hs = (unsigned short)r;
  __bf16 b;
  __builtin_memcpy(&b, &hs, 2);
  return b;
}

__device__ __forceinline__ float sigm(float x) { return 1.0f / (1.0f + expf(-x)); }

__device__ __forceinline__ unsigned char f32_to_fp8(float f) {
  unsigned char s = 0;
  if (f < 0.0f) { s = 0x80; f = -f; }
  if (!(f == f)) return 0x7F;
  if (f >= 448.0f) return (unsigned char)(s | 0x7E);
  if (f == 0.0f) return s;
  int e;
  (void)frexpf(f, &e);
  int E = e - 1;
  if (E < -6) {
    int qi = (int)rintf(ldexpf(f, 9));
    if (qi <= 0) return s;
    if (qi >= 8) return (unsigned char)(s | 0x08);
    return (unsigned char)(s | qi);
  }
  int qi = (int)rintf(ldexpf(f, -E) * 8.0f);
  if (qi == 16) { qi = 8; ++E; }
  if (E > 8) return (unsigned char)(s | 0x7E);
  unsigned char code = (unsigned char)(((E + 7) << 3) | (qi - 8));
  if (code == 0x7F) code = 0x7E;
  return (unsigned char)(s | code);
}

__device__ __forceinline__ float wave_sum(float v) {
  #pragma unroll
  for (int o = 16; o > 0; o >>= 1) v += __shfl_down(v, o);
  return v;
}
__device__ __forceinline__ float wave_max(float v) {
  #pragma unroll
  for (int o = 16; o > 0; o >>= 1) v = fmaxf(v, __shfl_down(v, o));
  return v;
}
__device__ __forceinline__ float block_sum(float v, float* sh) {
  __syncthreads();
  int lane = threadIdx.x & 31, wid = threadIdx.x >> 5, nw = (int)(blockDim.x >> 5);
  v = wave_sum(v);
  if (lane == 0) sh[wid] = v;
  __syncthreads();
  v = ((int)threadIdx.x < nw) ? sh[threadIdx.x] : 0.0f;
  if (wid == 0) { v = wave_sum(v); if (lane == 0) sh[0] = v; }
  __syncthreads();
  return sh[0];
}
__device__ __forceinline__ float block_max(float v, float* sh) {
  __syncthreads();
  int lane = threadIdx.x & 31, wid = threadIdx.x >> 5, nw = (int)(blockDim.x >> 5);
  v = wave_max(v);
  if (lane == 0) sh[wid] = v;
  __syncthreads();
  v = ((int)threadIdx.x < nw) ? sh[threadIdx.x] : -3.402823466e38f;
  if (wid == 0) { v = wave_max(v); if (lane == 0) sh[0] = v; }
  __syncthreads();
  return sh[0];
}


__global__ void init_state(const float* h0, const float* c0, float* h, float* c,
                           float* cost_acc, float* ym_sum) {
  int i = blockIdx.x * blockDim.x + threadIdx.x;
  if (i < B * H) { h[i] = h0[i]; c[i] = c0[i]; }
  if (i < B) { cost_acc[i] = 0.0f; ym_sum[i] = 0.0f; }
}

__global__ void add_bias_vec(const float* a, const float* b, float* o, int n) {
  int i = blockIdx.x * blockDim.x + threadIdx.x;
  if (i < n) o[i] = a[i] + b[i];
}

__global__ void pack_bf16_frag(__bf16* __restrict__ dst,
                               const float* __restrict__ s0, int k0,
                               const float* __restrict__ s1, int k1,
                               const float* __restrict__ s2, int k2,
                               int M, int K) {
  int total = M * K;
  int idx = blockIdx.x * blockDim.x + threadIdx.x;
  if (idx >= total) return;
  int e    = idx & 511;
  int f    = idx >> 9;
  int lane = e >> 4;
  int i    = e & 15;
  int ktiles = K >> 5;
  int mt = f / ktiles, kt = f - mt * ktiles;
  int row = mt * 16 + (lane & 15);
  int k   = kt * 32 + (i & 7) + ((i >> 3) << 4) + ((lane >> 4) << 3);
  float v;
  if (k < k0)            v = s0[(size_t)row * k0 + k];
  else if (k < k0 + k1)  v = s1[(size_t)row * k1 + (k - k0)];
  else                   v = s2[(size_t)row * k2 + (k - k0 - k1)];
  dst[idx] = to_bf16(v);
}

__global__ void gemm_bf16(const __bf16* __restrict__ Af, const __bf16* __restrict__ Wf,
                          const float* __restrict__ bias, float* __restrict__ C,
                          int K, int N, int Ntiles) {
  int wave = (int)(threadIdx.x >> 5);
  int lane = (int)(threadIdx.x & 31);
  int nt = blockIdx.x * (int)(blockDim.x >> 5) + wave;
  int mt = blockIdx.y;
  if (nt >= Ntiles) return;
  int ktiles = K >> 5;
  const __bf16* Ap = Af + ((size_t)mt * ktiles) * 512 + lane * 16;
  const __bf16* Bp = Wf + ((size_t)nt * ktiles) * 512 + lane * 16;
  v8f acc = {};
  for (int k = 0; k < ktiles; ++k) {
    v16bf a = *(const v16bf*)(Ap + (size_t)k * 512);
    v16bf b = *(const v16bf*)(Bp + (size_t)k * 512);
    acc = __builtin_amdgcn_wmma_f32_16x16x32_bf16(false, a, false, b,
                                                  (short)0, acc, false, false);
  }
  int n = nt * 16 + (lane & 15);
  int mbase = mt * 16 + ((lane >> 4) << 3);
  float bb = bias ? bias[n] : 0.0f;
  #pragma unroll
  for (int r = 0; r < 8; ++r)
    C[(size_t)(mbase + r) * N + n] = acc[r] + bb;
}


__global__ void lstm1_elem(const float* __restrict__ gates, const float* __restrict__ h,
                           const float* __restrict__ c, const float* __restrict__ ym,
                           float* __restrict__ h1, float* __restrict__ c1) {
  int i = blockIdx.x * blockDim.x + threadIdx.x;
  if (i >= B * H) return;
  int b = i >> 9, j = i & 511;
  const float* g = gates + (size_t)b * G4;
  float gi = g[j], gf = g[512 + j], gg = g[1024 + j], go = g[1536 + j];
  float cc = sigm(gf) * c[i] + sigm(gi) * tanhf(gg);
  float hh = sigm(go) * tanhf(cc);
  float m = ym[b];
  h1[i] = m * hh + (1.0f - m) * h[i];
  c1[i] = m * cc + (1.0f - m) * c[i];
}

__global__ void gate2_elem(const float* __restrict__ pre, const float* __restrict__ h1,
                           const float* __restrict__ c1, const float* __restrict__ ym,
                           float* __restrict__ h, float* __restrict__ c) {
  int i = blockIdx.x * blockDim.x + threadIdx.x;
  if (i >= B * H) return;
  int b = i >> 9, j = i & 511;
  const float* g = pre + (size_t)b * G4;
  float xi = g[j], xf = g[512 + j], xo = g[1024 + j], xc = g[1536 + j];
  float cc = sigm(xf) * c1[i] + sigm(xi) * tanhf(xc);
  float hh = sigm(xo) * tanhf(cc);
  float m = ym[b];
  c[i] = m * cc + (1.0f - m) * c1[i];
  h[i] = m * hh + (1.0f - m) * h1[i];
}


__global__ void att_scores(const float* __restrict__ pctx, const float* __restrict__ q,
                           const float* __restrict__ u_att, const float* __restrict__ x_mask,
                           float* __restrict__ u) {
  __shared__ float sh[8];
  int s = blockIdx.x, b = blockIdx.y;
  const float* pr = pctx + ((size_t)s * B + b) * CTX;
  const float* qr = q + (size_t)b * CTX;
  float sum = 0.0f;
  for (int cix = threadIdx.x; cix < CTX; cix += blockDim.x)
    sum += tanhf(pr[cix] + qr[cix]) * u_att[cix];
  sum = block_sum(sum, sh);
  if (threadIdx.x == 0) u[s * B + b] = sum * x_mask[s * B + b];
}

__global__ void softmax_att(const float* __restrict__ u, const float* __restrict__ x_mask,
                            float* __restrict__ w) {
  __shared__ float sh[8];
  int b = blockIdx.x;
  float mx = -3.402823466e38f;
  for (int s = threadIdx.x; s < S_LEN; s += blockDim.x)
    mx = fmaxf(mx, u[s * B + b]);
  mx = block_max(mx, sh);
  float sum = 0.0f;
  for (int s = threadIdx.x; s < S_LEN; s += blockDim.x) {
    float e = expf(u[s * B + b] - mx) * x_mask[s * B + b];
    w[s * B + b] = e;
    sum += e;
  }
  sum = block_sum(sum, sh);
  float inv = 1.0f / sum;
  for (int s = threadIdx.x; s < S_LEN; s += blockDim.x) w[s * B + b] *= inv;
}

__global__ void att_apply(const float* __restrict__ w, const float* __restrict__ context,
                          float* __restrict__ att) {
  int cix = blockIdx.x * blockDim.x + threadIdx.x;
  int b = blockIdx.y;
  if (cix >= CTX) return;
  float sum = 0.0f;
  for (int s = 0; s < S_LEN; ++s)
    sum += w[s * B + b] * context[((size_t)s * B + b) * CTX + cix];
  att[(size_t)b * CTX + cix] = sum;
}


__global__ void wp_rowscale(const float* __restrict__ Wp, float* __restrict__ wscale) {
  __shared__ float sh[8];
  int n = blockIdx.x;
  float mx = 0.0f;
  for (int k = threadIdx.x; k < G4; k += blockDim.x)
    mx = fmaxf(mx, fabsf(Wp[(size_t)n * G4 + k]));
  mx = block_max(mx, sh);
  if (threadIdx.x == 0) wscale[n] = fmaxf(mx, 1e-8f) * (1.0f / 448.0f);
}

__global__ void pack_wp_fp8(const float* __restrict__ Wp, const float* __restrict__ wscale,
                            unsigned char* __restrict__ dst) {
  size_t idx = (size_t)blockIdx.x * blockDim.x + threadIdx.x;
  if (idx >= (size_t)V * G4) return;
  size_t f = idx >> 11;
  int e    = (int)(idx & 2047);
  int lane = e >> 6;
  int j    = e & 63;
  int kt   = (int)(f & 15);
  int nt   = (int)(f >> 4);
  int n    = nt * 16 + (lane & 15);
  int k    = kt * 128 + ((j >> 4) << 5) + ((lane >> 4) << 4) + (j & 15);
  dst[idx] = f32_to_fp8(Wp[(size_t)n * G4 + k] / wscale[n]);
}

__global__ void feat_absmax(const float* __restrict__ h2, const float* __restrict__ att,
                            const float* __restrict__ y, float* __restrict__ ascale) {
  __shared__ float sh[8];
  int b = blockIdx.x;
  float mx = 0.0f;
  for (int k = threadIdx.x; k < G4; k += blockDim.x) {
    float v = (k < 512) ? h2[b * 512 + k]
            : (k < 1536) ? att[(size_t)b * CTX + (k - 512)]
            : y[b * 512 + (k - 1536)];
    mx = fmaxf(mx, fabsf(v));
  }
  mx = block_max(mx, sh);
  if (threadIdx.x == 0) ascale[b] = fmaxf(mx, 1e-8f) * (1.0f / 448.0f);
}

__global__ void pack_feat_fp8(const float* __restrict__ h2, const float* __restrict__ att,
                              const float* __restrict__ y, const float* __restrict__ ascale,
                              unsigned char* __restrict__ dst) {
  int idx = blockIdx.x * blockDim.x + threadIdx.x;
  if (idx >= B * G4) return;
  int f    = idx >> 11;
  int e    = idx & 2047;
  int lane = e >> 6;
  int j    = e & 63;
  int m    = lane & 15;
  int k    = f * 128 + ((j >> 4) << 5) + (j & 15)
           + ((lane >> 4) << 4);
  float v = (k < 512) ? h2[m * 512 + k]
          : (k < 1536) ? att[(size_t)m * CTX + (k - 512)]
          : y[m * 512 + (k - 1536)];
  dst[idx] = f32_to_fp8(v / ascale[m]);
}

__global__ void gemm_fp8_vocab(const unsigned char* __restrict__ Af,
                               const unsigned char* __restrict__ Wf,
                               const float* __restrict__ ascale,
                               const float* __restrict__ wscale,
                               const float* __restrict__ bias, float* __restrict__ C,
                               int K, int N, int Ntiles) {
  int wave = (int)(threadIdx.x >> 5);
  int lane = (int)(threadIdx.x & 31);
  int nt0 = (blockIdx.x * (int)(blockDim.x >> 5) + wave) * 2;
  if (nt0 >= Ntiles) return;
  int nt1 = (nt0 + 1 < Ntiles) ? nt0 + 1 : nt0;
  int ktiles = K >> 7;
  const unsigned char* Ap  = Af + lane * 64;
  const unsigned char* Bp0 = Wf + ((size_t)nt0 * ktiles) * 2048 + lane * 64;
  const unsigned char* Bp1 = Wf + ((size_t)nt1 * ktiles) * 2048 + lane * 64;
  v8f acc0 = {}, acc1 = {};
  for (int k = 0; k < ktiles; ++k) {
    v16i a  = *(const v16i*)(Ap  + (size_t)k * 2048);
    v16i b0 = *(const v16i*)(Bp0 + (size_t)k * 2048);
    v16i b1 = *(const v16i*)(Bp1 + (size_t)k * 2048);
    acc0 = __builtin_amdgcn_wmma_f32_16x16x128_fp8_fp8(a, b0, (short)0, acc0,
                                                       false, false);
    acc1 = __builtin_amdgcn_wmma_f32_16x16x128_fp8_fp8(a, b1, (short)0, acc1,
                                                       false, false);
  }
  int mbase = (lane >> 4) << 3;
  int n0 = nt0 * 16 + (lane & 15);
  float w0 = wscale[n0], bb0 = bias[n0];
  #pragma unroll
  for (int r = 0; r < 8; ++r) {
    int m = mbase + r;
    C[(size_t)m * N + n0] = acc0[r] * ascale[m] * w0 + bb0;
  }
  if (nt1 != nt0) {
    int n1 = nt1 * 16 + (lane & 15);
    float w1 = wscale[n1], bb1 = bias[n1];
    #pragma unroll
    for (int r = 0; r < 8; ++r) {
      int m = mbase + r;
      C[(size_t)m * N + n1] = acc1[r] * ascale[m] * w1 + bb1;
    }
  }
}

__global__ void nll_step(const float* __restrict__ logits, const int* __restrict__ yi,
                         const float* __restrict__ ym, float* __restrict__ cost_acc,
                         float* __restrict__ ym_sum) {
  __shared__ float sh[8];
  int b = blockIdx.x;
  const float* row = logits + (size_t)b * V;
  float mx = -3.402823466e38f;
  for (int i = threadIdx.x; i < V; i += blockDim.x) mx = fmaxf(mx, row[i]);
  mx = block_max(mx, sh);
  float sum = 0.0f;
  for (int i = threadIdx.x; i < V; i += blockDim.x) sum += expf(row[i] - mx);
  sum = block_sum(sum, sh);
  if (threadIdx.x == 0) {
    float lse = logf(sum) + mx;
    float m = ym[b];
    cost_acc[b] += m * (lse - row[yi[b]]);
    ym_sum[b]  += m;
  }
}

__global__ void finalize_cost(const float* cost_acc, const float* ym_sum, float* out) {
  int b = threadIdx.x;
  if (b < B) out[b] = cost_acc[b] / ym_sum[b];
}


extern "C" void kernel_launch(void* const* d_in, const int* in_sizes, int n_in,
                              void* d_out, int out_size, void* d_ws, size_t ws_size,
                              hipStream_t stream) {
  (void)in_sizes; (void)n_in; (void)out_size; (void)ws_size;
  const float* y_emb      = (const float*)d_in[0];
  const float* context    = (const float*)d_in[1];
  const float* init_h     = (const float*)d_in[2];
  const float* init_c     = (const float*)d_in[3];
  const float* x_mask     = (const float*)d_in[4];
  const float* y_mask     = (const float*)d_in[5];
  const int*   y_idx      = (const int*)  d_in[6];
  const float* W_ih       = (const float*)d_in[7];
  const float* W_hh       = (const float*)d_in[8];
  const float* b_ih       = (const float*)d_in[9];
  const float* b_hh       = (const float*)d_in[10];
  const float* Wx         = (const float*)d_in[11];
  const float* Ux         = (const float*)d_in[12];
  const float* bx         = (const float*)d_in[13];
  const float* Wc_att     = (const float*)d_in[14];
  const float* b_att      = (const float*)d_in[15];
  const float* W_comb_att = (const float*)d_in[16];
  const float* U_att      = (const float*)d_in[17];
  const float* Wp         = (const float*)d_in[18];
  const float* bp         = (const float*)d_in[19];
  float* out = (float*)d_out;

  char* base = (char*)d_ws;
  size_t off = 0;
  auto alloc = [&](size_t bytes) -> void* {
    void* p = base + off;
    off += (bytes + 255) & ~(size_t)255;
    return p;
  };

  __bf16* Wcat1f = (__bf16*)alloc((size_t)G4 * 1024 * 2);
  float*  bias1  = (float*) alloc((size_t)G4 * 4);
  __bf16* Wcombf = (__bf16*)alloc((size_t)CTX * 1024 * 2);
  __bf16* Wcat2f = (__bf16*)alloc((size_t)G4 * 1536 * 2);
  __bf16* WcAf   = (__bf16*)alloc((size_t)CTX * 1024 * 2);
  __bf16* CtxAf  = (__bf16*)alloc((size_t)S_LEN * B * CTX * 2);
  float*  pctx   = (float*) alloc((size_t)S_LEN * B * CTX * 4);
  unsigned char* Wp8 = (unsigned char*)alloc((size_t)V * G4);
  float*  wscale = (float*) alloc((size_t)V * 4);
  float*  gates  = (float*) alloc((size_t)B * G4 * 4);
  float*  h1     = (float*) alloc((size_t)B * H * 4);
  float*  c1     = (float*) alloc((size_t)B * H * 4);
  float*  hs     = (float*) alloc((size_t)B * H * 4);
  float*  cs     = (float*) alloc((size_t)B * H * 4);
  float*  qbuf   = (float*) alloc((size_t)B * CTX * 4);
  float*  ubuf   = (float*) alloc((size_t)S_LEN * B * 4);
  float*  wbuf   = (float*) alloc((size_t)S_LEN * B * 4);
  float*  attb   = (float*) alloc((size_t)B * CTX * 4);
  float*  preb   = (float*) alloc((size_t)B * G4 * 4);
  __bf16* A1f    = (__bf16*)alloc((size_t)B * 1024 * 2);
  __bf16* A2f    = (__bf16*)alloc((size_t)B * 1024 * 2);
  __bf16* A3f    = (__bf16*)alloc((size_t)B * 1536 * 2);
  unsigned char* feat8 = (unsigned char*)alloc((size_t)B * G4);
  float*  ascale = (float*) alloc(64);
  float*  logits = (float*) alloc((size_t)B * V * 4);
  float*  cacc   = (float*) alloc(64);
  float*  ymsum  = (float*) alloc(64);

  init_state<<<32, 256, 0, stream>>>(init_h, init_c, hs, cs, cacc, ymsum);
  add_bias_vec<<<8, 256, 0, stream>>>(b_ih, b_hh, bias1, G4);

  {
    int tot = G4 * 1024;
    pack_bf16_frag<<<(tot + 255) / 256, 256, 0, stream>>>(
        Wcat1f, W_ih, 512, W_hh, 512, nullptr, 0, G4, 1024);
  }
  {
    int tot = CTX * 1024;
    pack_bf16_frag<<<(tot + 255) / 256, 256, 0, stream>>>(
        Wcombf, W_comb_att, 1024, nullptr, 0, nullptr, 0, CTX, 1024);
  }
  {
    int tot = G4 * 1536;
    pack_bf16_frag<<<(tot + 255) / 256, 256, 0, stream>>>(
        Wcat2f, Ux, 512, Wx, 1024, nullptr, 0, G4, 1536);
  }
  {
    int tot = CTX * 1024;
    pack_bf16_frag<<<(tot + 255) / 256, 256, 0, stream>>>(
        WcAf, Wc_att, 1024, nullptr, 0, nullptr, 0, CTX, 1024);
  }
  {
    int tot = S_LEN * B * CTX;
    pack_bf16_frag<<<(tot + 255) / 256, 256, 0, stream>>>(
        CtxAf, context, CTX, nullptr, 0, nullptr, 0, S_LEN * B, CTX);
  }
  gemm_bf16<<<dim3(16, 400), 128, 0, stream>>>(CtxAf, WcAf, b_att, pctx,
                                               1024, CTX, CTX / 16);

  wp_rowscale<<<V, 256, 0, stream>>>(Wp, wscale);
  {
    size_t tot = (size_t)V * G4;
    pack_wp_fp8<<<(unsigned)((tot + 255) / 256), 256, 0, stream>>>(Wp, wscale, Wp8);
  }

  for (int t = 0; t < T_LEN; ++t) {
    const float* y_t  = y_emb  + (size_t)t * B * D_IN;
    const float* ym_t = y_mask + (size_t)t * B;
    const int*   yi_t = y_idx  + (size_t)t * B;

    pack_bf16_frag<<<64, 256, 0, stream>>>(A1f, y_t, 512, hs, 512, nullptr, 0, B, 1024);
    gemm_bf16<<<32, 128, 0, stream>>>(A1f, Wcat1f, bias1, gates, 1024, G4, G4 / 16);
    lstm1_elem<<<32, 256, 0, stream>>>(gates, hs, cs, ym_t, h1, c1);

    pack_bf16_frag<<<64, 256, 0, stream>>>(A2f, h1, 512, c1, 512, nullptr, 0, B, 1024);
    gemm_bf16<<<16, 128, 0, stream>>>(A2f, Wcombf, nullptr, qbuf, 1024, CTX, CTX / 16);

    att_scores<<<dim3(S_LEN, B), 256, 0, stream>>>(pctx, qbuf, U_att, x_mask, ubuf);
    softmax_att<<<B, 256, 0, stream>>>(ubuf, x_mask, wbuf);
    att_apply<<<dim3(4, B), 256, 0, stream>>>(wbuf, context, attb);

    pack_bf16_frag<<<96, 256, 0, stream>>>(A3f, h1, 512, attb, 1024, nullptr, 0, B, 1536);
    gemm_bf16<<<32, 128, 0, stream>>>(A3f, Wcat2f, bx, preb, 1536, G4, G4 / 16);
    gate2_elem<<<32, 256, 0, stream>>>(preb, h1, c1, ym_t, hs, cs);

    feat_absmax<<<B, 256, 0, stream>>>(hs, attb, y_t, ascale);
    pack_feat_fp8<<<128, 256, 0, stream>>>(hs, attb, y_t, ascale, feat8);
    gemm_fp8_vocab<<<(V / 16 + 7) / 8, 128, 0, stream>>>(
        feat8, Wp8, ascale, wscale, bp, logits, G4, V, V / 16);
    nll_step<<<B, 256, 0, stream>>>(logits, yi_t, ym_t, cacc, ymsum);
  }

  finalize_cost<<<1, 32, 0, stream>>>(cacc, ymsum, out);
}
